// QuantizedAttention_69776038691248
// MI455X (gfx1250) — matrix operand order repaired and checked by reading, not compiled, not run on hardware
//
#include <hip/hip_runtime.h>

// ---------------------------------------------------------------------------
// MX-quantized top-k attention for gfx1250 (MI455X).
// All matmuls run on v_wmma_f32_16x16x32_bf16: MXINT8 values (int8 mantissa x
// power-of-2 shared scale, block=32) are exactly representable in bf16, so a
// bf16 WMMA with f32 accumulation reproduces the MX matmul semantics.
// The big GEMMs stage tiles in LDS with GLOBAL_LOAD_ASYNC_TO_LDS_B128
// (ASYNCcnt-tracked, double buffered, s_wait_asynccnt).
// ---------------------------------------------------------------------------

typedef __bf16 bf16_t;
typedef bf16_t v16bf __attribute__((ext_vector_type(16)));
typedef float  v8f   __attribute__((ext_vector_type(8)));
typedef unsigned int v4u __attribute__((ext_vector_type(4)));

union Frag {
    v16bf v;
    v4u   q[2];
    unsigned short u[16];
};

// ---------------- scalar helpers -------------------------------------------

__device__ __forceinline__ float bf_round(float f) {
    // round-to-nearest-even to bf16 grid, keep f32 container
    unsigned u = __float_as_uint(f);
    u += 0x7FFFu + ((u >> 16) & 1u);
    return __uint_as_float(u & 0xFFFF0000u);
}
__device__ __forceinline__ unsigned short f2bf(float f) {
    unsigned u = __float_as_uint(f);
    u += 0x7FFFu + ((u >> 16) & 1u);
    return (unsigned short)(u >> 16);
}
__device__ __forceinline__ float bf2f(unsigned short h) {
    return __uint_as_float(((unsigned)h) << 16);
}
__device__ __forceinline__ float expsign(float x) {
    // sign(x) * 2^floor(log2|x|); 0 -> 0
    if (x == 0.0f) return 0.0f;
    return __uint_as_float(__float_as_uint(x) & 0xFF800000u);
}
// MXINT8 quantize one (already bf16-rounded) element given its block amax.
// Result is exactly representable in bf16.
__device__ __forceinline__ unsigned short mxq(float v, float amax) {
    float p2 = (amax > 0.0f)
                 ? __uint_as_float(__float_as_uint(amax) & 0x7F800000u)
                 : 1.0f;                       // 2^floor(log2 amax)
    float scale = p2 * 0.015625f;              // 2^(shared_exp - 6)
    float r = rintf(v / scale);                // RNE like jnp.round
    r = fminf(127.0f, fmaxf(-127.0f, r));
    return f2bf(r * scale);
}
__device__ __forceinline__ float wave_max(float v) {
    #pragma unroll
    for (int o = 16; o; o >>= 1) v = fmaxf(v, __shfl_xor(v, o));
    return v;
}

// ---------------- gfx1250 async copy helpers --------------------------------
// Per-lane async copy of 16B global -> LDS, tracked by ASYNCcnt.
// VDST operand is the LDS byte address (low 32 bits of the generic pointer;
// ISA 10.2: flat LDS addresses truncate to the DS offset).
__device__ __forceinline__ void async_b128(void* lds, const void* gaddr) {
    unsigned l = (unsigned)(size_t)lds;
    asm volatile("global_load_async_to_lds_b128 %0, %1, off"
                 :: "v"(l), "v"(gaddr)
                 : "memory");
}
template <int NN>
__device__ __forceinline__ void wait_async() {
    asm volatile("s_wait_asynccnt %0" :: "n"(NN) : "memory");
}

// ---------------- WMMA fragment loaders ------------------------------------
// A fragment (16x32 bf16, row-major src, leading dim ld in elements):
//   lane L: m = L&15, half = L>>4; elem i -> k = 16*half + i  (contiguous)
__device__ __forceinline__ v16bf frag_A(const unsigned short* base, int ld, int lane) {
    const int m = lane & 15, half = lane >> 4;
    Frag f;
    const unsigned short* p = base + m * ld + 16 * half;
    f.q[0] = *(const v4u*)(p);
    f.q[1] = *(const v4u*)(p + 8);
    return f.v;
}
// B fragment (32x16 bf16) where B[k][n] = src[n*ld + k] (i.e. src row-major
// [N, K], as for weights / K-matrix):
//   lane L: n = L&15, half = L>>4; elem i -> k = 16*half + i  (contiguous)
__device__ __forceinline__ v16bf frag_B(const unsigned short* base, int ld, int lane) {
    const int n = lane & 15, half = lane >> 4;
    Frag f;
    const unsigned short* p = base + n * ld + 16 * half;
    f.q[0] = *(const v4u*)(p);
    f.q[1] = *(const v4u*)(p + 8);
    return f.v;
}
// B fragment from a [K, N] row-major matrix (column-strided gather), colbase
// already points at (k0, n): elem i -> value at row (16*half + i).
__device__ __forceinline__ v16bf frag_B_strided(const unsigned short* colbase, int ld, int lane) {
    Frag f;
    const int half = lane >> 4;
    const unsigned short* p = colbase + 16 * half * ld;
    #pragma unroll
    for (int i = 0; i < 16; i++) f.u[i] = p[i * ld];
    return f.v;
}
__device__ __forceinline__ v8f wmma_bf16(v16bf a, v16bf b, v8f c) {
    return __builtin_amdgcn_wmma_f32_16x16x32_bf16(false, a, false, b,
                                                   (short)0, c, false, false);
}

// ---------------- kernels ---------------------------------------------------

// bf16-round + MX quant along last axis, blocks of 32 (rows are multiples of
// 32, so one wave == one block on a flat index).
__global__ __launch_bounds__(256) void quant_rows_kernel(
    const float* __restrict__ src, unsigned short* __restrict__ dst) {
    int t = blockIdx.x * 256 + threadIdx.x;
    float v = bf_round(src[t]);
    float am = wave_max(fabsf(v));
    dst[t] = mxq(v, am);
}

// out[M,N] = A[M,K](bf16) @ W[N,K]^T(bf16) + bias[N], f32 accumulate.
// 8 waves per block; block tile 32(M) x 64(N), K staged in 64-wide LDS tiles
// via async global->LDS copies, double buffered. Each wave owns a 16x16 tile
// and runs two WMMAs per stage.
__global__ __launch_bounds__(256) void gemm_bf16_kernel(
    const unsigned short* __restrict__ A, const unsigned short* __restrict__ W,
    const float* __restrict__ bias, float* __restrict__ out,
    int M, int N, int K) {
    __shared__ __align__(16) unsigned short tA[2][32 * 64];  // 2 x 4KB
    __shared__ __align__(16) unsigned short tB[2][64 * 64];  // 2 x 8KB
    const int tid  = threadIdx.x;
    const int lane = tid & 31;
    const int wid  = tid >> 5;
    const int wm   = wid >> 2;        // 0..1
    const int wn   = wid & 3;         // 0..3
    const int row0 = blockIdx.y * 32;
    const int col0 = blockIdx.x * 64;
    const int lr = tid >> 3;          // 0..31
    const int lc = (tid & 7) * 8;     // 0,8,..,56

    const unsigned short* gA  = A + (size_t)(row0 + lr) * K + lc;
    const unsigned short* gB0 = W + (size_t)(col0 + lr) * K + lc;
    const unsigned short* gB1 = W + (size_t)(col0 + lr + 32) * K + lc;

    auto issue = [&](int k0, int buf) {
        async_b128(&tA[buf][lr * 64 + lc], gA + k0);
        async_b128(&tB[buf][lr * 64 + lc], gB0 + k0);
        async_b128(&tB[buf][(lr + 32) * 64 + lc], gB1 + k0);
    };

    v8f acc = {};
    const int nk = K >> 6;            // K / 64
    issue(0, 0);
    for (int kt = 0; kt < nk; kt++) {
        const int cur = kt & 1;
        if (kt + 1 < nk) {
            issue((kt + 1) << 6, cur ^ 1);
            wait_async<3>();          // previous stage's 3 copies complete
        } else {
            wait_async<0>();
        }
        __syncthreads();
        const unsigned short* at = &tA[cur][wm * 16 * 64];
        const unsigned short* wt = &tB[cur][wn * 16 * 64];
        v16bf a0 = frag_A(at, 64, lane);
        v16bf a1 = frag_A(at + 32, 64, lane);
        v16bf b0 = frag_B(wt, 64, lane);
        v16bf b1 = frag_B(wt + 32, 64, lane);
        acc = wmma_bf16(a0, b0, acc);
        acc = wmma_bf16(a1, b1, acc);
        __syncthreads();              // done reading before buffer reuse
    }
    const int nloc = lane & 15, halfl = lane >> 4;
    const int n = col0 + wn * 16 + nloc;
    const float bv = bias[n];
    #pragma unroll
    for (int r = 0; r < 8; r++) {
        int m = row0 + wm * 16 + r + 8 * halfl;
        out[(size_t)m * N + n] = acc[r] + bv;
    }
}

// One wave per (b,h,token): produce MX-quantized q*Dh^-0.5 and k along Dh
// (two 32-blocks), plus exponent-sign eq/ek, all as bf16 in [B,H,N,Dh].
__global__ __launch_bounds__(256) void split_qkv_kernel(
    const float* __restrict__ qkv, unsigned short* __restrict__ qs,
    unsigned short* __restrict__ ks, unsigned short* __restrict__ eqb,
    unsigned short* __restrict__ ekb) {
    int tid = blockIdx.x * 256 + threadIdx.x;
    int lane = tid & 31;
    int gw = tid >> 5;                 // 0 .. B*H*N-1
    int n = gw & 1023;
    int h = (gw >> 10) % 12;
    int b = gw / (1024 * 12);
    const float* row = qkv + ((size_t)(b * 1024 + n)) * 2304;
    float q0 = row[h * 64 + lane] * 0.125f;           // Dh^-0.5 = 0.125
    float q1 = row[h * 64 + lane + 32] * 0.125f;
    float k0 = row[768 + h * 64 + lane];
    float k1 = row[768 + h * 64 + lane + 32];
    size_t o = (((size_t)(b * 12 + h)) * 1024 + n) * 64;
    eqb[o + lane]      = f2bf(expsign(q0));
    eqb[o + lane + 32] = f2bf(expsign(q1));
    ekb[o + lane]      = f2bf(expsign(k0));
    ekb[o + lane + 32] = f2bf(expsign(k1));
    q0 = bf_round(q0); q1 = bf_round(q1);
    k0 = bf_round(k0); k1 = bf_round(k1);
    float aq0 = wave_max(fabsf(q0)), aq1 = wave_max(fabsf(q1));
    float ak0 = wave_max(fabsf(k0)), ak1 = wave_max(fabsf(k1));
    qs[o + lane]      = mxq(q0, aq0);
    qs[o + lane + 32] = mxq(q1, aq1);
    ks[o + lane]      = mxq(k0, ak0);
    ks[o + lane + 32] = mxq(k1, ak1);
}

// v quantized along the token axis (reference: axis=-2), blocks of 32 tokens
// per column; output [B,H,N,Dh] bf16.
__global__ __launch_bounds__(256) void quant_v_kernel(
    const float* __restrict__ qkv, unsigned short* __restrict__ vs) {
    int t = blockIdx.x * 256 + threadIdx.x;   // 0 .. B*H*32*64-1
    int d  = t & 63;
    int nb = (t >> 6) & 31;
    int h  = (t >> 11) % 12;
    int b  = t / (2048 * 12);
    float vals[32];
    float am = 0.0f;
    #pragma unroll 4
    for (int i = 0; i < 32; i++) {
        float v = bf_round(qkv[((size_t)(b * 1024 + nb * 32 + i)) * 2304 +
                               1536 + h * 64 + d]);
        vals[i] = v;
        am = fmaxf(am, fabsf(v));
    }
    size_t o = (((size_t)(b * 12 + h)) * 1024 + nb * 32) * 64 + d;
    #pragma unroll 4
    for (int i = 0; i < 32; i++) vs[o + (size_t)i * 64] = mxq(vals[i], am);
}

// Fused attention: one wave per (b, h, 16-query tile).
//  - S = q@k^T and P = eq@ek^T via bf16 WMMA into LDS rows (bf16)
//  - per-row top-20 on P (low-index tie-break, matches lax.top_k)
//  - masked softmax over selected S entries, MX re-quant of prob rows
//  - out = P_q @ v via bf16 WMMA (32 K-steps x 4 N-tiles)
__global__ __launch_bounds__(32) void attn_topk_kernel(
    const unsigned short* __restrict__ qs, const unsigned short* __restrict__ ks,
    const unsigned short* __restrict__ vs, const unsigned short* __restrict__ eqb,
    const unsigned short* __restrict__ ekb, float* __restrict__ out) {
    __shared__ __align__(16) unsigned short sbuf[16 * 1024];  // scores / probs
    __shared__ __align__(16) unsigned short pbuf[16 * 1024];  // predictor
    const int lane = threadIdx.x;
    const int gid = blockIdx.x;
    const int rt = gid & 63;
    const int h  = (gid >> 6) % 12;
    const int b  = gid / (64 * 12);
    const size_t head = ((size_t)(b * 12 + h)) * (1024 * 64);
    const unsigned short* Q  = qs  + head + rt * 16 * 64;
    const unsigned short* EQ = eqb + head + rt * 16 * 64;
    const unsigned short* Km = ks  + head;
    const unsigned short* Em = ekb + head;
    const unsigned short* Vm = vs  + head;
    const int nloc = lane & 15, halfl = lane >> 4;

    v16bf aq0 = frag_A(Q, 64, lane);
    v16bf aq1 = frag_A(Q + 32, 64, lane);
    v16bf ae0 = frag_A(EQ, 64, lane);
    v16bf ae1 = frag_A(EQ + 32, 64, lane);

    for (int j = 0; j < 64; j++) {
        const unsigned short* kb = Km + j * 16 * 64;
        const unsigned short* eb = Em + j * 16 * 64;
        v8f s = {};
        s = wmma_bf16(aq0, frag_B(kb, 64, lane), s);
        s = wmma_bf16(aq1, frag_B(kb + 32, 64, lane), s);
        v8f p = {};
        p = wmma_bf16(ae0, frag_B(eb, 64, lane), p);
        p = wmma_bf16(ae1, frag_B(eb + 32, 64, lane), p);
        #pragma unroll
        for (int r = 0; r < 8; r++) {
            int m = r + 8 * halfl;
            sbuf[m * 1024 + j * 16 + nloc] = f2bf(s[r]);
            pbuf[m * 1024 + j * 16 + nloc] = f2bf(p[r]);
        }
    }
    __syncthreads();

    for (int row = 0; row < 16; row++) {
        unsigned short* sr = sbuf + row * 1024;
        unsigned short* pr = pbuf + row * 1024;
        float svals[20];
        int   sidx[20];
        float rowmax = -1e30f;
        for (int t = 0; t < 20; t++) {
            float bv = -1e38f;
            int   bi = 1 << 30;
            for (int c = lane; c < 1024; c += 32) {
                float v = bf2f(pr[c]);
                if (v > bv || (v == bv && c < bi)) { bv = v; bi = c; }
            }
            #pragma unroll
            for (int o = 16; o; o >>= 1) {
                float ov = __shfl_xor(bv, o);
                int   oi = __shfl_xor(bi, o);
                if (ov > bv || (ov == bv && oi < bi)) { bv = ov; bi = oi; }
            }
            if (lane == 0) pr[bi] = 0xFF80;   // bf16 -inf: remove from pool
            float sv = bf2f(sr[bi]);
            svals[t] = sv;
            sidx[t]  = bi;
            rowmax = fmaxf(rowmax, sv);
            __syncthreads();
        }
        float sum = 0.0f;
        for (int t = 0; t < 20; t++) sum += __expf(svals[t] - rowmax);
        float inv = 1.0f / sum;
        for (int c = lane; c < 1024; c += 32) sr[c] = 0;  // non-selected -> 0
        __syncthreads();
        if (lane == 0) {
            for (int t = 0; t < 20; t++)
                sr[sidx[t]] = f2bf(__expf(svals[t] - rowmax) * inv);
        }
        __syncthreads();
        // MX quant of the prob row: lane owns one 32-block
        {
            const int b0 = lane * 32;
            float am = 0.0f;
            for (int i = 0; i < 32; i++) am = fmaxf(am, fabsf(bf2f(sr[b0 + i])));
            for (int i = 0; i < 32; i++) sr[b0 + i] = mxq(bf2f(sr[b0 + i]), am);
        }
        __syncthreads();
    }

    // out[16,64] = probs[16,1024] @ v[1024,64]
    v8f acc[4] = {};
    for (int kt = 0; kt < 32; kt++) {
        v16bf a = frag_A(sbuf + kt * 32, 1024, lane);
        const unsigned short* vb = Vm + kt * 32 * 64;
        #pragma unroll
        for (int nt = 0; nt < 4; nt++) {
            v16bf bfr = frag_B_strided(vb + nt * 16 + nloc, 64, lane);
            acc[nt] = wmma_bf16(a, bfr, acc[nt]);
        }
    }
    #pragma unroll
    for (int nt = 0; nt < 4; nt++) {
        #pragma unroll
        for (int r = 0; r < 8; r++) {
            int m = r + 8 * halfl;
            int tok = rt * 16 + m;
            out[((size_t)(b * 1024 + tok)) * 768 + h * 64 + nt * 16 + nloc] =
                acc[nt][r];
        }
    }
}

// ---------------- host launcher ---------------------------------------------

extern "C" void kernel_launch(void* const* d_in, const int* in_sizes, int n_in,
                              void* d_out, int out_size, void* d_ws, size_t ws_size,
                              hipStream_t stream) {
    (void)in_sizes; (void)n_in; (void)out_size; (void)ws_size;
    const float* x      = (const float*)d_in[0];
    const float* qkv_w  = (const float*)d_in[1];
    const float* qkv_b  = (const float*)d_in[2];
    const float* proj_w = (const float*)d_in[3];
    const float* proj_b = (const float*)d_in[4];
    float* out = (float*)d_out;

    char* ws = (char*)d_ws;
    size_t off = 0;
    auto carve = [&](size_t bytes) -> char* {
        char* p = ws + off;
        off += (bytes + 255) & ~(size_t)255;
        return p;
    };
    float*          qkvbuf = (float*)carve((size_t)4096 * 2304 * 4);
    unsigned short* xq     = (unsigned short*)carve((size_t)4096 * 768 * 2);
    unsigned short* wq     = (unsigned short*)carve((size_t)2304 * 768 * 2);
    unsigned short* pwq    = (unsigned short*)carve((size_t)768 * 768 * 2);
    unsigned short* qsb    = (unsigned short*)carve((size_t)48 * 1024 * 64 * 2);
    unsigned short* ksb    = (unsigned short*)carve((size_t)48 * 1024 * 64 * 2);
    unsigned short* vsb    = (unsigned short*)carve((size_t)48 * 1024 * 64 * 2);
    unsigned short* eqb    = (unsigned short*)carve((size_t)48 * 1024 * 64 * 2);
    unsigned short* ekb    = (unsigned short*)carve((size_t)48 * 1024 * 64 * 2);
    float*          ao     = (float*)carve((size_t)4096 * 768 * 4);
    unsigned short* aoq    = (unsigned short*)carve((size_t)4096 * 768 * 2);

    // 1) quantize activations + qkv weights
    quant_rows_kernel<<<12288, 256, 0, stream>>>(x, xq);          // 4096*768
    quant_rows_kernel<<<6912, 256, 0, stream>>>(qkv_w, wq);       // 2304*768
    // 2) QKV GEMM
    gemm_bf16_kernel<<<dim3(2304 / 64, 4096 / 32), 256, 0, stream>>>(
        xq, wq, qkv_b, qkvbuf, 4096, 2304, 768);
    // 3) split + per-head quantization
    split_qkv_kernel<<<6144, 256, 0, stream>>>(qkvbuf, qsb, ksb, eqb, ekb);
    quant_v_kernel<<<384, 256, 0, stream>>>(qkvbuf, vsb);
    // 4) fused top-k attention
    attn_topk_kernel<<<3072, 32, 0, stream>>>(qsb, ksb, vsb, eqb, ekb, ao);
    // 5) projection
    quant_rows_kernel<<<12288, 256, 0, stream>>>(ao, aoq);        // 4096*768
    quant_rows_kernel<<<2304, 256, 0, stream>>>(proj_w, pwq);     // 768*768
    gemm_bf16_kernel<<<dim3(768 / 64, 4096 / 32), 256, 0, stream>>>(
        aoq, pwq, proj_b, out, 4096, 768, 768);
}
